// SelfAttention_48859547959606
// MI455X (gfx1250) — matrix operand order repaired and checked by reading, not compiled, not run on hardware
//
#include <hip/hip_runtime.h>

// Problem constants (from reference): B=4, C=256, Dqk=32, H=W=64, N=4096
#define NB 4
#define CC 256
#define DD 32
#define NN 4096

typedef __attribute__((ext_vector_type(16))) __bf16 v16bf;
typedef __attribute__((ext_vector_type(8)))  __bf16 v8bf;
typedef __attribute__((ext_vector_type(8)))  float  v8f;

// xor-16 lane swap across the wave32 (group-of-32 ds_swizzle: SWAPX16)
__device__ __forceinline__ float swapx16(float v) {
    return __int_as_float(
        __builtin_amdgcn_ds_swizzle(__float_as_int(v), 0x401f));
}

// ---------------------------------------------------------------------------
// Kernel 1: fused 1x1-conv QKV projection, fp32 compute, bf16 outputs laid out
// for WMMA operand loading:
//   qT[b][n][d]  (row n contiguous over d)   -> Q B-operand (32B/lane)
//   kT[b][n][d]                              -> K A-tiles (2x16B/lane)
//   vb[b][c][n]  (row c contiguous over n)   -> V A-tiles (2x16B/lane)
// ---------------------------------------------------------------------------
__global__ void __launch_bounds__(256)
qkv_kernel(const float* __restrict__ x,
           const float* __restrict__ Wq, const float* __restrict__ bq,
           const float* __restrict__ Wk, const float* __restrict__ bk,
           const float* __restrict__ Wv, const float* __restrict__ bv,
           void* __restrict__ ws)
{
    __bf16* qT = (__bf16*)ws;
    __bf16* kT = qT + (size_t)NB * NN * DD;
    __bf16* vb = kT + (size_t)NB * NN * DD;

    __shared__ float xt[CC * 32];                 // 32 KB x tile [c][n]
    const int b  = blockIdx.x >> 7;               // N/32 = 128 tiles per batch
    const int n0 = (blockIdx.x & 127) << 5;
    const int t  = threadIdx.x;

    for (int idx = t; idx < CC * 32; idx += 256) {
        xt[idx] = x[((size_t)(b * CC + (idx >> 5))) * NN + n0 + (idx & 31)];
    }
    __syncthreads();

    // ---- V rows: one output channel per thread, 32 positions each ----
    {
        float acc[32];
        #pragma unroll
        for (int n = 0; n < 32; ++n) acc[n] = 0.0f;
        const float* wr = Wv + t * CC;
        for (int c = 0; c < CC; ++c) {
            const float w = wr[c];
            #pragma unroll
            for (int n = 0; n < 32; ++n) acc[n] = fmaf(w, xt[c * 32 + n], acc[n]);
        }
        const float bias = bv[t];
        #pragma unroll
        for (int n = 0; n < 32; ++n)
            vb[((size_t)(b * CC + t)) * NN + n0 + n] = (__bf16)(acc[n] + bias);
    }

    // ---- Q / K rows: threads 0..63 ----
    if (t < 64) {
        const int d        = t & 31;
        const float* wr    = ((t < 32) ? Wq : Wk) + d * CC;
        const float  bias  = (t < 32) ? bq[d] : bk[d];
        __bf16*      dst   = (t < 32) ? qT : kT;
        float acc[32];
        #pragma unroll
        for (int n = 0; n < 32; ++n) acc[n] = 0.0f;
        for (int c = 0; c < CC; ++c) {
            const float w = wr[c];
            #pragma unroll
            for (int n = 0; n < 32; ++n) acc[n] = fmaf(w, xt[c * 32 + n], acc[n]);
        }
        #pragma unroll
        for (int n = 0; n < 32; ++n)
            dst[((size_t)(b * NN + n0 + n)) * DD + d] = (__bf16)(acc[n] + bias);
    }
}

// ---------------------------------------------------------------------------
// Kernel 2: flash-attention, transposed-S formulation, zero LDS.
// Block = 128 threads = 4 independent waves; each wave owns 16 query columns
// (i = i0+ln per lane) and the full C=256 output as 16 x v8f accumulators in
// [c-rows][i-cols] layout.
//
// Per 32-wide j-step:
//   2x  v_wmma : S'^T tiles = K_A x Q_B       (D-layout: lane = column i)
//   softmax stats: 15 fmax / 15 add local + ONE ds_swizzle(SWAPX16) each
//   P pack: register-local (A-tile j-row remap makes D-layout == B-layout)
//   16x v_wmma : O'[c][i] += V_A x P_B
// Epilogue: coalesced 64B row segments, out = x + gamma * O / l.
// ---------------------------------------------------------------------------
__global__ void __launch_bounds__(128, 1)
attn_kernel(const float* __restrict__ x,
            const float* __restrict__ gamma,
            const void* __restrict__ ws,
            float* __restrict__ out)
{
    const __bf16* qT = (const __bf16*)ws;
    const __bf16* kT = qT + (size_t)NB * NN * DD;
    const __bf16* vb = kT + (size_t)NB * NN * DD;

    const int wave = threadIdx.x >> 5;
    const int lane = threadIdx.x & 31;
    const int ln   = lane & 15;                 // query column i (local)
    const int h    = lane >> 4;                 // lane half
    const int b    = blockIdx.x >> 6;           // 64 i-blocks per batch
    const int i0   = ((blockIdx.x & 63) << 6) + (wave << 4);

    // Q as B operand (32d x 16i), loop invariant: half h holds d = 16h+e, i = ln
    const v16bf bQ = *(const v16bf*)(qT + ((size_t)(b * NN) + i0 + ln) * DD + 16 * h);

    v8f accO[16];
    #pragma unroll
    for (int ct = 0; ct < 16; ++ct) accO[ct] = (v8f){};
    float mrow = -1e30f, lrow = 0.0f;

    // K A-tile row for this lane: tile0 row m=ln  <->  j = j0 + (ln&7) + 16*(ln>>3)
    // With this remap, D-layout p-values land exactly in B-operand order.
    const int jrow = (ln & 7) + 16 * (ln >> 3);
    const __bf16* kbase = kT + (size_t)b * NN * DD;
    const __bf16* vbase = vb + (size_t)b * CC * NN;

    for (int j0 = 0; j0 < NN; j0 += 32) {
        // K A-tiles (16j x 32d): elems 0..15 = d 16h..16h+15
        const __bf16* k0 = kbase + (size_t)(j0 + jrow) * DD + 16 * h;
        const __bf16* k1 = k0 + 8 * DD;         // tile1: j + 8
        v16bf aK0, aK1;
        {
            v8bf a0 = *(const v8bf*)(k0);
            v8bf a1 = *(const v8bf*)(k0 + 8);
            v8bf a2 = *(const v8bf*)(k1);
            v8bf a3 = *(const v8bf*)(k1 + 8);
            #pragma unroll
            for (int e = 0; e < 8; ++e) {
                aK0[e] = a0[e]; aK0[8 + e] = a1[e];
                aK1[e] = a2[e]; aK1[8 + e] = a3[e];
            }
        }
        const v8f z = (v8f){};
        v8f S0 = __builtin_amdgcn_wmma_f32_16x16x32_bf16(
            false, aK0, false, bQ, (short)0, z, false, false);
        v8f S1 = __builtin_amdgcn_wmma_f32_16x16x32_bf16(
            false, aK1, false, bQ, (short)0, z, false, false);
        // lane holds S'[j0+16h+r][i=ln] (S0) and S'[j0+16h+8+r][i=ln] (S1)

        // ---- online softmax over j for column i=ln ----
        float cm = fmaxf(S0[0], S1[0]);
        #pragma unroll
        for (int r = 1; r < 8; ++r) cm = fmaxf(cm, fmaxf(S0[r], S1[r]));
        cm = fmaxf(cm, swapx16(cm));            // other 16 j live in partner half
        const float mnew = fmaxf(mrow, cm);
        const bool  grew = mnew > mrow;
        const float scl  = __expf(mrow - mnew);

        #pragma unroll
        for (int r = 0; r < 8; ++r) {
            S0[r] = __expf(S0[r] - mnew);
            S1[r] = __expf(S1[r] - mnew);
        }
        float s = S0[0] + S1[0];
        #pragma unroll
        for (int r = 1; r < 8; ++r) s += S0[r] + S1[r];
        s += swapx16(s);
        lrow = lrow * scl + s;
        mrow = mnew;

        // rescale O only when the running max actually moved (wave-uniform)
        if (__any(grew)) {
            #pragma unroll
            for (int ct = 0; ct < 16; ++ct)
                #pragma unroll
                for (int r = 0; r < 8; ++r) accO[ct][r] *= scl;
        }

        // P as B operand (32j x 16i): pure register pack, no data movement
        v16bf bP;
        #pragma unroll
        for (int r = 0; r < 8; ++r) {
            bP[r]     = (__bf16)S0[r];
            bP[8 + r] = (__bf16)S1[r];
        }

        // ---- O'[c][i] += V_A x P_B over all 16 channel tiles ----
        #pragma unroll
        for (int ct = 0; ct < 16; ++ct) {
            const __bf16* vp = vbase + (size_t)(ct * 16 + ln) * NN + j0 + 16 * h;
            v8bf v0 = *(const v8bf*)(vp);
            v8bf v1 = *(const v8bf*)(vp + 8);
            v16bf aV;
            #pragma unroll
            for (int e = 0; e < 8; ++e) { aV[e] = v0[e]; aV[8 + e] = v1[e]; }
            accO[ct] = __builtin_amdgcn_wmma_f32_16x16x32_bf16(
                false, aV, false, bP, (short)0, accO[ct], false, false);
        }
    }

    // ---- epilogue: out = x + gamma * (O / l), 64B-coalesced per (ct,r) ----
    const float g    = gamma[0];
    const float invl = 1.0f / lrow;
    #pragma unroll
    for (int ct = 0; ct < 16; ++ct) {
        #pragma unroll
        for (int r = 0; r < 8; ++r) {
            const int    c   = ct * 16 + r + 8 * h;
            const size_t idx = ((size_t)(b * CC + c)) * NN + i0 + ln;
            out[idx] = x[idx] + g * (accO[ct][r] * invl);
        }
    }
}

// ---------------------------------------------------------------------------
extern "C" void kernel_launch(void* const* d_in, const int* in_sizes, int n_in,
                              void* d_out, int out_size, void* d_ws, size_t ws_size,
                              hipStream_t stream)
{
    (void)in_sizes; (void)n_in; (void)out_size; (void)ws_size;
    const float* x     = (const float*)d_in[0];
    const float* Wq    = (const float*)d_in[1];
    const float* bq    = (const float*)d_in[2];
    const float* Wk    = (const float*)d_in[3];
    const float* bk    = (const float*)d_in[4];
    const float* Wv    = (const float*)d_in[5];
    const float* bv    = (const float*)d_in[6];
    const float* gamma = (const float*)d_in[7];
    float*       out   = (float*)d_out;

    // Workspace: qT (1MB) + kT (1MB) + v_bf16 (8MB) = 10MB bf16.
    qkv_kernel<<<NB * (NN / 32), 256, 0, stream>>>(x, Wq, bq, Wk, bk, Wv, bv, d_ws);
    attn_kernel<<<NB * (NN / 64), 128, 0, stream>>>(x, gamma, d_ws, out);
}
